// MambaLayer_29850022708037
// MI455X (gfx1250) — compile-verified, matrix operand order checked by reading, not run on hardware
//
#include <hip/hip_runtime.h>
#include <hip/hip_bf16.h>

// ---------------------------------------------------------------------------
// Mamba layer for MI455X (gfx1250, wave32, WMMA).
// Dims: B=1, C=128, L=32768, D_INNER=256, D_STATE=16, DT_RANK=8, D_CONV=4.
// ---------------------------------------------------------------------------

#define LSEQ   32768
#define CDIM   128
#define DINNER 256
#define NSTATE 16
#define DTRANK 8
#define DCONV  4
#define CHUNK  128              // scan chunk length
#define NCHUNK (LSEQ / CHUNK)   // 256

typedef _Float16 v8h  __attribute__((ext_vector_type(8)));
typedef _Float16 v16h __attribute__((ext_vector_type(16)));
typedef float    v8f  __attribute__((ext_vector_type(8)));

static inline size_t align_up(size_t x) { return (x + 255) & ~(size_t)255; }
static inline int cdiv(int a, int b) { return (a + b - 1) / b; }

// ---------------------------------------------------------------------------
// Fused LayerNorm over C, emit f16 activations (L x 128 row-major).
// x is (C, L): element (l, c) at x[c*L + l]  -> fully coalesced across l.
// ---------------------------------------------------------------------------
__global__ void ln_kernel(const float* __restrict__ x,
                          const float* __restrict__ w,
                          const float* __restrict__ b,
                          _Float16* __restrict__ xn) {
  int l = blockIdx.x * blockDim.x + threadIdx.x;
  if (l >= LSEQ) return;
  float s = 0.f, ss = 0.f;
  for (int c = 0; c < CDIM; ++c) {
    float v = x[(size_t)c * LSEQ + l];
    s += v; ss += v * v;
  }
  float mu   = s * (1.0f / CDIM);
  float var  = ss * (1.0f / CDIM) - mu * mu;
  float rstd = rsqrtf(var + 1e-5f);
  _Float16* row = xn + (size_t)l * CDIM;
  for (int c = 0; c < CDIM; ++c) {
    float v = x[(size_t)c * LSEQ + l];
    row[c] = (_Float16)((v - mu) * rstd * w[c] + b[c]);
  }
}

// ---------------------------------------------------------------------------
// Pack row-major fp32 weight W (Nout x Kin) into per-lane WMMA B fragments
// (f16), matching the 16-bit 16x32 operand layout:
//   lane<16 : n=lane,    k in {base..base+7, base+16..base+23}, base=0
//   lane>=16: n=lane-16, same with base=8
// Packed layout: [jt][kt][lane*16 + i], 512 halfs per tile.
// Out-of-range (j,k) -> 0 (free N/K padding).
// ---------------------------------------------------------------------------
__global__ void pack_b_kernel(const float* __restrict__ W,
                              _Float16* __restrict__ Bp,
                              int Nout, int Kin, int JT, int KT) {
  int idx = blockIdx.x * blockDim.x + threadIdx.x;
  int total = JT * KT * 512;
  if (idx >= total) return;
  int jt   = idx / (KT * 512);
  int r    = idx % (KT * 512);
  int kt   = r >> 9;
  int q    = r & 511;
  int lane = q >> 4;
  int i    = q & 15;
  int n    = lane & 15;
  int vg   = i >> 1, hh = i & 1;
  int base8 = (lane >> 4) << 3;
  int kk = base8 + ((vg < 4) ? (2 * vg + hh) : (16 + 2 * (vg - 4) + hh));
  int j = jt * 16 + n;
  int k = kt * 32 + kk;
  float v = (j < Nout && k < Kin) ? W[(size_t)j * Kin + k] : 0.f;
  Bp[idx] = (_Float16)v;
}

// ---------------------------------------------------------------------------
// WMMA GEMM: each wave computes a 16 x (NT*16) strip of C.
// A fragment is loaded once per k-step and reused across NT tiles; the
// K-loop is double-buffered (next k-step's loads issued before this step's
// WMMAs) so v_wmma overlaps outstanding global_load_b128.
// ---------------------------------------------------------------------------
template <int NT>
__global__ void gemm_wmma_f16(const _Float16* __restrict__ A, int lda,
                              const _Float16* __restrict__ Bp,
                              float* __restrict__ Cf,
                              _Float16* __restrict__ Ch, int ldc,
                              int Mtiles, int NGroups, int Ktiles) {
  int wave = (int)((blockIdx.x * blockDim.x + threadIdx.x) >> 5);
  int lane = threadIdx.x & 31;
  if (wave >= Mtiles * NGroups) return;        // wave-uniform
  int tm = wave / NGroups, tg = wave % NGroups;
  int m0  = tm << 4;
  int tn0 = tg * NT;

  int mrow = m0 + (lane & 15);
  int koff = (lane >> 4) << 3;                 // 0 or 8
  const _Float16* arow = A + (size_t)mrow * lda + koff;
  const _Float16* bb[NT];
  #pragma unroll
  for (int j = 0; j < NT; ++j)
    bb[j] = Bp + (((size_t)(tn0 + j) * Ktiles) << 9) + (lane << 4);

  auto loadA = [&](int kt) -> v16h {
    const _Float16* ap = arow + (kt << 5);
    v8h lo = *(const v8h*)(ap);
    v8h hi = *(const v8h*)(ap + 16);
    return __builtin_shufflevector(lo, hi,
        0, 1, 2, 3, 4, 5, 6, 7, 8, 9, 10, 11, 12, 13, 14, 15);
  };

  v8f acc[NT];
  #pragma unroll
  for (int j = 0; j < NT; ++j) acc[j] = (v8f){};

  // prologue: fragments for kt = 0
  v16h a_cur = loadA(0);
  v16h b_cur[NT];
  #pragma unroll
  for (int j = 0; j < NT; ++j) b_cur[j] = *(const v16h*)(bb[j]);

  for (int kt = 0; kt < Ktiles - 1; ++kt) {
    // issue next k-step's loads first (overlap with WMMAs below)
    v16h a_nxt = loadA(kt + 1);
    v16h b_nxt[NT];
    #pragma unroll
    for (int j = 0; j < NT; ++j)
      b_nxt[j] = *(const v16h*)(bb[j] + ((kt + 1) << 9));
    #pragma unroll
    for (int j = 0; j < NT; ++j)
      acc[j] = __builtin_amdgcn_wmma_f32_16x16x32_f16(
          false, a_cur, false, b_cur[j], (short)0, acc[j], false, false);
    a_cur = a_nxt;
    #pragma unroll
    for (int j = 0; j < NT; ++j) b_cur[j] = b_nxt[j];
  }
  #pragma unroll
  for (int j = 0; j < NT; ++j)
    acc[j] = __builtin_amdgcn_wmma_f32_16x16x32_f16(
        false, a_cur, false, b_cur[j], (short)0, acc[j], false, false);

  int mstore = m0 + ((lane >> 4) << 3);        // rows 0..7 or 8..15
  int nlane  = lane & 15;
  #pragma unroll
  for (int j = 0; j < NT; ++j) {
    int ncol = ((tn0 + j) << 4) + nlane;
    #pragma unroll
    for (int r = 0; r < 8; ++r) {
      size_t off = (size_t)(mstore + r) * ldc + ncol;
      if (Cf) Cf[off] = acc[j][r];
      if (Ch) Ch[off] = (_Float16)acc[j][r];
    }
  }
}

// ---------------------------------------------------------------------------
// Depthwise causal conv (k=4) along L on xm (= xz cols 0..255), + bias +
// SiLU. Output u f16 (L x 256).
// ---------------------------------------------------------------------------
__global__ void conv_silu_kernel(const _Float16* __restrict__ xz,
                                 const float* __restrict__ cw,
                                 const float* __restrict__ cb,
                                 _Float16* __restrict__ u) {
  int t = blockIdx.x * blockDim.x + threadIdx.x;
  if (t >= LSEQ * DINNER) return;
  int d = t & (DINNER - 1);
  int l = t >> 8;
  float acc = cb[d];
  #pragma unroll
  for (int tt = 0; tt < DCONV; ++tt) {
    int ll = l - (DCONV - 1) + tt;
    if (ll >= 0)
      acc += cw[d * DCONV + tt] * (float)xz[(size_t)ll * 512 + d];
  }
  float sig = 1.0f / (1.0f + __expf(-acc));
  u[t] = (_Float16)(acc * sig);
}

// ---------------------------------------------------------------------------
// delta = softplus(dt_raw + dt_bias)
// ---------------------------------------------------------------------------
__global__ void softplus_kernel(const _Float16* __restrict__ dtraw,
                                const float* __restrict__ bias,
                                _Float16* __restrict__ delta) {
  int t = blockIdx.x * blockDim.x + threadIdx.x;
  if (t >= LSEQ * DINNER) return;
  int d = t & (DINNER - 1);
  float v = (float)dtraw[t] + bias[d];
  float r = (v > 20.f) ? v : log1pf(__expf(v));
  delta[t] = (_Float16)r;
}

// ---------------------------------------------------------------------------
// Scan pass 1: per (chunk g, d, n) compute decay product P and zero-state
// chunk output h_out.  a = exp(delta*A), b = delta*u*Bm.
// ---------------------------------------------------------------------------
__global__ void scan_pass1(const _Float16* __restrict__ delta,
                           const _Float16* __restrict__ u,
                           const _Float16* __restrict__ dbl,
                           const float* __restrict__ A_log,
                           float* __restrict__ P,
                           float* __restrict__ Hout) {
  int tid = blockIdx.x * blockDim.x + threadIdx.x;
  if (tid >= NCHUNK * DINNER * NSTATE) return;
  int n = tid & 15;
  int d = (tid >> 4) & 255;
  int g = tid >> 12;
  float Aval = -__expf(A_log[d * NSTATE + n]);
  float p = 1.f, h = 0.f;
  int l0 = g * CHUNK;
  for (int i = 0; i < CHUNK; ++i) {
    int l = l0 + i;
    float dlt = (float)delta[(size_t)l * DINNER + d];
    float uu  = (float)u[(size_t)l * DINNER + d];
    float bb  = (float)dbl[(size_t)l * 48 + DTRANK + n];
    float a   = __expf(dlt * Aval);
    p *= a;
    h = a * h + dlt * uu * bb;
  }
  int dn = d * NSTATE + n;
  P[(size_t)g * (DINNER * NSTATE) + dn]    = p;
  Hout[(size_t)g * (DINNER * NSTATE) + dn] = h;
}

// ---------------------------------------------------------------------------
// Scan pass 2: sequential combine over the 256 chunks (4096 threads).
// ---------------------------------------------------------------------------
__global__ void scan_pass2(const float* __restrict__ P,
                           const float* __restrict__ Hout,
                           float* __restrict__ Hin) {
  int dn = blockIdx.x * blockDim.x + threadIdx.x;
  if (dn >= DINNER * NSTATE) return;
  float h = 0.f;
  for (int g = 0; g < NCHUNK; ++g) {
    size_t off = (size_t)g * (DINNER * NSTATE) + dn;
    Hin[off] = h;
    h = P[off] * h + Hout[off];
  }
}

// ---------------------------------------------------------------------------
// Scan pass 3: rescan each chunk with the correct carry-in; fuse the
// n-reduction (shfl_xor over 16-lane groups), y += u*D_skip, y *= silu(z),
// write yfin f16 (L x 256) — the A matrix of out_proj.
// ---------------------------------------------------------------------------
__global__ void scan_pass3(const _Float16* __restrict__ delta,
                           const _Float16* __restrict__ u,
                           const _Float16* __restrict__ dbl,
                           const _Float16* __restrict__ xz,   // z = cols 256..511
                           const float* __restrict__ A_log,
                           const float* __restrict__ Dskip,
                           const float* __restrict__ Hin,
                           _Float16* __restrict__ yfin) {
  int tid = blockIdx.x * blockDim.x + threadIdx.x;
  if (tid >= NCHUNK * DINNER * NSTATE) return;
  int n = tid & 15;
  int d = (tid >> 4) & 255;
  int g = tid >> 12;
  float Aval = -__expf(A_log[d * NSTATE + n]);
  float h = Hin[(size_t)g * (DINNER * NSTATE) + d * NSTATE + n];
  int l0 = g * CHUNK;
  for (int i = 0; i < CHUNK; ++i) {
    int l = l0 + i;
    float dlt = (float)delta[(size_t)l * DINNER + d];
    float uu  = (float)u[(size_t)l * DINNER + d];
    float bb  = (float)dbl[(size_t)l * 48 + DTRANK + n];
    float cc  = (float)dbl[(size_t)l * 48 + DTRANK + NSTATE + n];
    float a   = __expf(dlt * Aval);
    h = a * h + dlt * uu * bb;
    float yc = h * cc;
    yc += __shfl_xor(yc, 1, 16);
    yc += __shfl_xor(yc, 2, 16);
    yc += __shfl_xor(yc, 4, 16);
    yc += __shfl_xor(yc, 8, 16);
    if (n == 0) {
      float yv = yc + uu * Dskip[d];
      float zz = (float)xz[(size_t)l * 512 + DINNER + d];
      yv *= zz / (1.0f + __expf(-zz));
      yfin[(size_t)l * DINNER + d] = (_Float16)yv;
    }
  }
}

// ---------------------------------------------------------------------------
// Final permute: out (L x 128 f32) -> (C, Dd, W, H)-ordered d_out
// (reference swaps the last two spatial axes). Coalesced writes.
// ---------------------------------------------------------------------------
__global__ void permute_out_kernel(const float* __restrict__ outbuf,
                                   float* __restrict__ dst) {
  int o = blockIdx.x * blockDim.x + threadIdx.x;
  if (o >= CDIM * LSEQ) return;
  int c  = o >> 15;
  int dd = (o >> 10) & 31;
  int a  = (o >> 5) & 31;
  int b  = o & 31;
  int l  = dd * 1024 + b * 32 + a;
  dst[o] = outbuf[(size_t)l * CDIM + c];
}

// ---------------------------------------------------------------------------
extern "C" void kernel_launch(void* const* d_in, const int* in_sizes, int n_in,
                              void* d_out, int out_size, void* d_ws, size_t ws_size,
                              hipStream_t stream) {
  const float* x      = (const float*)d_in[0];
  const float* ln_w   = (const float*)d_in[1];
  const float* ln_b   = (const float*)d_in[2];
  const float* inp_w  = (const float*)d_in[3];   // (512,128)
  const float* conv_w = (const float*)d_in[4];   // (256,1,4)
  const float* conv_b = (const float*)d_in[5];
  const float* xprj_w = (const float*)d_in[6];   // (40,256)
  const float* dt_w   = (const float*)d_in[7];   // (256,8)
  const float* dt_b   = (const float*)d_in[8];
  const float* A_log  = (const float*)d_in[9];   // (256,16)
  const float* D_skip = (const float*)d_in[10];
  const float* out_w  = (const float*)d_in[11];  // (128,256)
  float* out = (float*)d_out;

  // ---- workspace layout -------------------------------------------------
  char* p = (char*)d_ws;
  auto take = [&](size_t bytes) { char* r = p; p += align_up(bytes); return r; };
  _Float16* xn16    = (_Float16*)take((size_t)LSEQ * CDIM   * 2);
  _Float16* xz16    = (_Float16*)take((size_t)LSEQ * 512    * 2);
  _Float16* u16     = (_Float16*)take((size_t)LSEQ * DINNER * 2);
  _Float16* dbl16   = (_Float16*)take((size_t)LSEQ * 48     * 2);
  _Float16* dtraw16 = (_Float16*)take((size_t)LSEQ * DINNER * 2);
  _Float16* delta16 = (_Float16*)take((size_t)LSEQ * DINNER * 2);
  _Float16* yfin16  = (_Float16*)take((size_t)LSEQ * DINNER * 2);
  float*    outbuf  = (float*)   take((size_t)LSEQ * CDIM   * 4);
  float*    Pbuf    = (float*)   take((size_t)NCHUNK * DINNER * NSTATE * 4);
  float*    Hout    = (float*)   take((size_t)NCHUNK * DINNER * NSTATE * 4);
  float*    Hin     = (float*)   take((size_t)NCHUNK * DINNER * NSTATE * 4);
  _Float16* wpA     = (_Float16*)take((size_t)32 * 4 * 512 * 2);  // in_proj  512x128
  _Float16* wpX     = (_Float16*)take((size_t)3  * 8 * 512 * 2);  // x_proj   48(pad)x256
  _Float16* wpD     = (_Float16*)take((size_t)16 * 1 * 512 * 2);  // dt_proj  256x32(pad)
  _Float16* wpO     = (_Float16*)take((size_t)8  * 8 * 512 * 2);  // out_proj 128x256
  (void)ws_size; (void)in_sizes; (void)n_in; (void)out_size;

  const int B = 256;

  // ---- pack weights into WMMA B fragments -------------------------------
  pack_b_kernel<<<cdiv(32 * 4 * 512, B), B, 0, stream>>>(inp_w,  wpA, 512, 128, 32, 4);
  pack_b_kernel<<<cdiv(3  * 8 * 512, B), B, 0, stream>>>(xprj_w, wpX, 40,  256, 3,  8);
  pack_b_kernel<<<cdiv(16 * 1 * 512, B), B, 0, stream>>>(dt_w,   wpD, 256, 8,   16, 1);
  pack_b_kernel<<<cdiv(8  * 8 * 512, B), B, 0, stream>>>(out_w,  wpO, 128, 256, 8,  8);

  // ---- LayerNorm -> xn f16 ----------------------------------------------
  ln_kernel<<<cdiv(LSEQ, B), B, 0, stream>>>(x, ln_w, ln_b, xn16);

  // ---- in_proj: (32768x128) -> 512   (NT=4, 8 n-groups) ------------------
  {
    int Mt = LSEQ / 16, NG = (512 / 16) / 4;
    gemm_wmma_f16<4><<<(Mt * NG * 32) / B, B, 0, stream>>>(
        xn16, CDIM, wpA, (float*)nullptr, xz16, 512, Mt, NG, 4);
  }

  // ---- depthwise causal conv + SiLU -> u f16 ----------------------------
  conv_silu_kernel<<<cdiv(LSEQ * DINNER, B), B, 0, stream>>>(xz16, conv_w, conv_b, u16);

  // ---- x_proj: (32768x256) -> 40 (padded 48; NT=3, 1 group) --------------
  {
    int Mt = LSEQ / 16, NG = 1;
    gemm_wmma_f16<3><<<(Mt * NG * 32) / B, B, 0, stream>>>(
        u16, DINNER, wpX, (float*)nullptr, dbl16, 48, Mt, NG, 8);
  }

  // ---- dt_proj: (32768x8 from dbl cols 0..7, K padded 32) -> 256 ---------
  {
    int Mt = LSEQ / 16, NG = (DINNER / 16) / 4;
    gemm_wmma_f16<4><<<(Mt * NG * 32) / B, B, 0, stream>>>(
        dbl16, 48, wpD, (float*)nullptr, dtraw16, DINNER, Mt, NG, 1);
  }
  softplus_kernel<<<cdiv(LSEQ * DINNER, B), B, 0, stream>>>(dtraw16, dt_b, delta16);

  // ---- chunked selective scan -------------------------------------------
  scan_pass1<<<cdiv(NCHUNK * DINNER * NSTATE, B), B, 0, stream>>>(
      delta16, u16, dbl16, A_log, Pbuf, Hout);
  scan_pass2<<<cdiv(DINNER * NSTATE, B), B, 0, stream>>>(Pbuf, Hout, Hin);
  scan_pass3<<<cdiv(NCHUNK * DINNER * NSTATE, B), B, 0, stream>>>(
      delta16, u16, dbl16, xz16, A_log, D_skip, Hin, yfin16);

  // ---- out_proj: (32768x256) -> 128   (NT=4, 2 n-groups) -----------------
  {
    int Mt = LSEQ / 16, NG = (CDIM / 16) / 4;
    gemm_wmma_f16<4><<<(Mt * NG * 32) / B, B, 0, stream>>>(
        yfin16, DINNER, wpO, outbuf, (_Float16*)nullptr, CDIM, Mt, NG, 8);
  }

  // ---- final permute into d_out -----------------------------------------
  permute_out_kernel<<<cdiv(CDIM * LSEQ, B), B, 0, stream>>>(outbuf, out);
}
